// MambaLayer_14224931685047
// MI455X (gfx1250) — compile-verified, matrix operand order checked by reading, not run on hardware
//
#include <hip/hip_runtime.h>
#include <hip/hip_bf16.h>
#include <math.h>

// ---------------------------------------------------------------------------
// Mamba layer for MI455X (gfx1250, wave32).
//  k0 : one-time f32->f16 weight conversion (in_proj / x_proj / out_proj)
//  k1 : LayerNorm + in_proj (WMMA f16; A hoisted, B fully preloaded)
//  k2 : depthwise causal conv(K=4) + SiLU         -> uc (f16)
//  k3 : x_proj (WMMA, 4-deep B pipeline) + fused dt_proj + softplus
//  k5a: chunk-local selective scan (T=128)        -> h_local, sum(dt)
//  k5b: per-channel carry scan (mult = exp(A*sum_dt))
//  k5c: chunk re-scan with carry-in, y=(scan+uc*D)*silu(z) -> y (f16)
//  k6 : out_proj (WMMA, 4-deep B pipeline)        -> d_out (f32)
// ---------------------------------------------------------------------------

typedef _Float16 f16;
typedef __attribute__((ext_vector_type(16))) _Float16 v16h;
typedef __attribute__((ext_vector_type(8)))  _Float16 v8h;
typedef __attribute__((ext_vector_type(8)))  float    v8f;

#define BB     2
#define LL     16384
#define DMODEL 192
#define DINC   384
#define NST    16
#define KCONV  4
#define RTR    12
#define NTOK   (BB * LL)        // 32768 tokens
#define TCH    128              // scan chunk length
#define NCH    (LL / TCH)       // 128 chunks per sequence

// Fast SiLU: one v_rcp_f32 instead of an IEEE div_scale/div_fmas chain.
__device__ __forceinline__ float siluf(float x) {
  return x * __builtin_amdgcn_rcpf(1.0f + __expf(-x));
}

// A fragment (16xK tile, row-major f16 source with leading dim `ld`).
// ISA 7.12.2: lane m=lane&15; lane[4] selects K {0..7,16..23} vs {8..15,24..31}.
__device__ __forceinline__ v16h load_a(const f16* __restrict__ base, int ld,
                                       int lane, int kBase) {
  const int m    = lane & 15;
  const int koff = (lane >> 4) << 3;
  const f16* row = base + m * ld + kBase + koff;
  v16h a;
#pragma unroll
  for (int i = 0; i < 8; ++i) { a[i] = row[i]; a[8 + i] = row[16 + i]; }
  return a;
}

// B fragment from an f16 weight stored [E_out, K] row-major: two b128 loads.
__device__ __forceinline__ v16h load_bh(const f16* __restrict__ w, int ldw,
                                        int eBase, int kBase, int lane, int eMax) {
  const int n    = lane & 15;
  const int koff = (lane >> 4) << 3;
  const int e    = eBase + n;
  v16h b;
  if (e < eMax) {
    const f16* row = w + (size_t)e * ldw + kBase + koff;
#pragma unroll
    for (int i = 0; i < 8; ++i) { b[i] = row[i]; b[8 + i] = row[16 + i]; }
  } else {
#pragma unroll
    for (int i = 0; i < 16; ++i) b[i] = (f16)0.f;
  }
  return b;
}

// ---------------- k0: f32 -> f16 weight conversion ----------------
__global__ __launch_bounds__(256) void k_cvt(const float* __restrict__ src,
                                             f16* __restrict__ dst, int n) {
  const int i = blockIdx.x * 256 + threadIdx.x;
  if (i < n) dst[i] = (f16)src[i];
}

// ---------------- k1: LayerNorm + in_proj ----------------
__global__ __launch_bounds__(256) void k_ln_inproj(
    const float* __restrict__ x, const float* __restrict__ nw,
    const float* __restrict__ nb, const f16* __restrict__ w_in,
    f16* __restrict__ u_out, f16* __restrict__ zsil_out) {
  __shared__ f16 sA[16][200];  // 16 tokens x 192 (padded)
  const int tid  = threadIdx.x;
  const int lane = tid & 31;
  const int wave = tid >> 5;
  const int tok0 = blockIdx.x * 16;

  __builtin_prefetch(w_in + (size_t)tid * 64, 0, 3);  // global_prefetch_b8
  __builtin_amdgcn_s_wait_tensorcnt(0);

  // Phase A: LN for 16 tokens; 16 lanes per token (2 tokens per wave32).
  {
    const int tl = tid >> 4;
    const int j  = tid & 15;
    const float* row = x + (size_t)(tok0 + tl) * DMODEL;
    float v[12], s = 0.f, ss = 0.f;
#pragma unroll
    for (int i = 0; i < 12; ++i) { v[i] = row[j * 12 + i]; s += v[i]; ss += v[i] * v[i]; }
#pragma unroll
    for (int off = 8; off >= 1; off >>= 1) { s += __shfl_xor(s, off); ss += __shfl_xor(ss, off); }
    const float mean = s * (1.0f / DMODEL);
    const float var  = ss * (1.0f / DMODEL) - mean * mean;
    const float rstd = rsqrtf(var + 1e-5f);
#pragma unroll
    for (int i = 0; i < 12; ++i) {
      const int k = j * 12 + i;
      sA[tl][k] = (f16)(((v[i] - mean) * rstd) * nw[k] + nb[k]);
    }
  }
  __syncthreads();

  // Hoist A fragments (reused by every N-tile this wave computes).
  v16h afr[6];
#pragma unroll
  for (int kk = 0; kk < 6; ++kk) afr[kk] = load_a(&sA[0][0], 200, lane, kk * 32);

  // Phase B: 48 output tiles of 16; all 6 B-frags preloaded (distinct regs)
  // so the 12 b128 loads issue as one clause and WMMAs drain with
  // descending loadcnt waits instead of wait-0 serialization.
  for (int tile = wave; tile < 48; tile += 8) {
    const int eBase = tile * 16;
    v16h bq[6];
#pragma unroll
    for (int kk = 0; kk < 6; ++kk)
      bq[kk] = load_bh(w_in, DMODEL, eBase, kk * 32, lane, 2 * DINC);
    v8f acc = {};
#pragma unroll
    for (int kk = 0; kk < 6; ++kk)
      acc = __builtin_amdgcn_wmma_f32_16x16x32_f16(false, afr[kk], false, bq[kk],
                                                   (short)0, acc, false, false);
    const int n  = lane & 15;
    const int mB = (lane >> 4) << 3;
    const int e  = eBase + n;
    if (e < DINC) {
#pragma unroll
      for (int r = 0; r < 8; ++r)
        u_out[(size_t)(tok0 + mB + r) * DINC + e] = (f16)acc[r];
    } else {
#pragma unroll
      for (int r = 0; r < 8; ++r)
        zsil_out[(size_t)(tok0 + mB + r) * DINC + (e - DINC)] = (f16)siluf(acc[r]);
    }
  }
}

// ---------------- k2: causal depthwise conv + SiLU ----------------
__global__ __launch_bounds__(256) void k_conv_silu(
    const f16* __restrict__ u, const float* __restrict__ cw,
    const float* __restrict__ cb, f16* __restrict__ uc) {
  const size_t idx = (size_t)blockIdx.x * 256 + threadIdx.x;
  if (idx >= (size_t)NTOK * DINC) return;
  const int d   = (int)(idx % DINC);
  const int tok = (int)(idx / DINC);
  const int b = tok / LL, t = tok % LL;
  float acc = cb[d];
#pragma unroll
  for (int k = 0; k < KCONV; ++k) {
    const int tt = t - (KCONV - 1) + k;
    if (tt >= 0) acc += (float)u[(size_t)(b * LL + tt) * DINC + d] * cw[d * KCONV + k];
  }
  uc[idx] = (f16)siluf(acc);
}

// ---------------- k3: x_proj (WMMA) + fused dt_proj + softplus ----------------
__global__ __launch_bounds__(256) void k_xproj_dt(
    const f16* __restrict__ uc, const f16* __restrict__ w_x,
    const float* __restrict__ wdt, const float* __restrict__ bdt,
    float* __restrict__ dbc, float* __restrict__ dt) {
  __shared__ f16   sU[16][392];
  __shared__ float sD[16][48];
  const int tid = threadIdx.x, lane = tid & 31, wave = tid >> 5;
  const int tok0 = blockIdx.x * 16;
  __builtin_prefetch(w_x + (size_t)tid * 64, 0, 3);

  // Stage uc tile (16 x 384 f16) with 16B vector copies.
  {
    const v8h* src = (const v8h*)(uc + (size_t)tok0 * DINC);  // 48 v8h per row
    for (int i = tid; i < 16 * 48; i += 256) {
      const int m = i / 48, kb = i % 48;
      *(v8h*)&sU[m][kb * 8] = src[m * 48 + kb];
    }
  }
  __syncthreads();

  // x_proj: 3 N-tiles (44 outputs padded to 48), K=384 -> 12 WMMA steps.
  // 4-deep rotating B queue: each reload has ~3 WMMA latencies to land.
  if (wave < 3) {
    v16h afr[12];
#pragma unroll
    for (int kk = 0; kk < 12; ++kk) afr[kk] = load_a(&sU[0][0], 392, lane, kk * 32);
    const int eBase = wave * 16;
    v16h bq[4];
#pragma unroll
    for (int i = 0; i < 4; ++i)
      bq[i] = load_bh(w_x, DINC, eBase, i * 32, lane, RTR + 2 * NST);
    v8f acc = {};
#pragma unroll
    for (int kk = 0; kk < 12; ++kk) {
      acc = __builtin_amdgcn_wmma_f32_16x16x32_f16(false, afr[kk], false, bq[kk & 3],
                                                   (short)0, acc, false, false);
      if (kk + 4 < 12)
        bq[kk & 3] = load_bh(w_x, DINC, eBase, (kk + 4) * 32, lane, RTR + 2 * NST);
    }
    const int n = lane & 15, mB = (lane >> 4) << 3, e = eBase + n;
#pragma unroll
    for (int r = 0; r < 8; ++r) {
      sD[mB + r][e] = acc[r];
      dbc[(size_t)(tok0 + mB + r) * 48 + e] = acc[r];  // Bm/Cm for the scan
    }
  }
  __syncthreads();

  // Fused dt_proj (K=12) + softplus straight from LDS; d fastest -> coalesced.
  for (int i = tid; i < 16 * DINC; i += 256) {
    const int tl = i / DINC, d = i % DINC;
    float acc = bdt[d];
#pragma unroll
    for (int rr = 0; rr < RTR; ++rr) acc += sD[tl][rr] * wdt[d * RTR + rr];
    dt[(size_t)(tok0 + tl) * DINC + d] =
        (acc > 20.f) ? acc : __logf(1.0f + __expf(acc));
  }
}

// ---------------- k5a: chunk-local scan (h0 = 0) ----------------
// Wave32 = 2 channels x 16 states.  prod(exp(dt*A)) == exp(A*sum(dt)).
__global__ __launch_bounds__(256) void k_scan_local(
    const float* __restrict__ dt, const f16* __restrict__ uc,
    const float* __restrict__ dbc, const float* __restrict__ A_log,
    float* __restrict__ hloc, float* __restrict__ dtsum) {
  const int tid = threadIdx.x, lane = tid & 31, wave = tid >> 5;
  const int dblk = blockIdx.x % (DINC / 16);
  const int c    = (blockIdx.x / (DINC / 16)) % NCH;
  const int b    = blockIdx.x / ((DINC / 16) * NCH);
  const int n = lane & 15;
  const int d = dblk * 16 + wave * 2 + (lane >> 4);
  const float A = -__expf(A_log[d * NST + n]);
  float h = 0.f, S = 0.f;
  const int t0 = c * TCH;
  for (int t = 0; t < TCH; ++t) {
    const size_t tok = (size_t)b * LL + t0 + t;
    const float dtv = dt[tok * DINC + d];
    const float uv  = (float)uc[tok * DINC + d];
    const float bn  = dbc[tok * 48 + RTR + n];
    h = h * __expf(dtv * A) + dtv * uv * bn;
    S += dtv;
  }
  const size_t ch = (size_t)(b * DINC + d) * NCH + c;
  hloc[ch * NST + n] = h;
  if (n == 0) dtsum[ch] = S;
}

// ---------------- k5b: carry scan over 128 chunks per channel ----------------
__global__ __launch_bounds__(256) void k_scan_carry(
    const float* __restrict__ hloc, const float* __restrict__ dtsum,
    const float* __restrict__ A_log, float* __restrict__ h0) {
  const int tid = threadIdx.x, lane = tid & 31, wave = tid >> 5;
  const int gd = blockIdx.x * 16 + wave * 2 + (lane >> 4);  // 0..B*DINC-1
  const int n = lane & 15;
  const int d = gd % DINC;
  const float A = -__expf(A_log[d * NST + n]);
  float h = 0.f;
  const size_t base = (size_t)gd * NCH;
  for (int c = 0; c < NCH; ++c) {
    h0[(base + c) * NST + n] = h;  // carry-in for chunk c
    h = hloc[(base + c) * NST + n] + __expf(A * dtsum[base + c]) * h;
  }
}

// ---------------- k5c: final chunk scan, y = (scan + uc*D)*silu(z) ----------------
__global__ __launch_bounds__(256) void k_scan_final(
    const float* __restrict__ dt, const f16* __restrict__ uc,
    const float* __restrict__ dbc, const float* __restrict__ A_log,
    const float* __restrict__ Dskip, const f16* __restrict__ zsil,
    const float* __restrict__ h0, f16* __restrict__ y) {
  const int tid = threadIdx.x, lane = tid & 31, wave = tid >> 5;
  const int dblk = blockIdx.x % (DINC / 16);
  const int c    = (blockIdx.x / (DINC / 16)) % NCH;
  const int b    = blockIdx.x / ((DINC / 16) * NCH);
  const int n = lane & 15;
  const int d = dblk * 16 + wave * 2 + (lane >> 4);
  const float A  = -__expf(A_log[d * NST + n]);
  const float Dd = Dskip[d];
  const size_t ch = (size_t)(b * DINC + d) * NCH + c;
  float h = h0[ch * NST + n];
  const int t0 = c * TCH;
  for (int t = 0; t < TCH; ++t) {
    const size_t tok = (size_t)b * LL + t0 + t;
    const float dtv = dt[tok * DINC + d];
    const float uv  = (float)uc[tok * DINC + d];
    const float bn  = dbc[tok * 48 + RTR + n];
    const float cn  = dbc[tok * 48 + RTR + NST + n];
    h = h * __expf(dtv * A) + dtv * uv * bn;
    float p = h * cn;
#pragma unroll
    for (int off = 8; off >= 1; off >>= 1) p += __shfl_xor(p, off);  // 16-lane sum
    const float zs = (float)zsil[tok * DINC + d];
    const float yv = (p + uv * Dd) * zs;
    if (n == 0) y[tok * DINC + d] = (f16)yv;
  }
}

// ---------------- k6: out_proj ----------------
__global__ __launch_bounds__(256) void k_outproj(
    const f16* __restrict__ y, const f16* __restrict__ w_out,
    float* __restrict__ out) {
  __shared__ f16 sY[16][392];
  const int tid = threadIdx.x, lane = tid & 31, wave = tid >> 5;
  const int tok0 = blockIdx.x * 16;
  __builtin_prefetch(w_out + (size_t)tid * 64, 0, 3);
  {
    const v8h* src = (const v8h*)(y + (size_t)tok0 * DINC);
    for (int i = tid; i < 16 * 48; i += 256) {
      const int m = i / 48, kb = i % 48;
      *(v8h*)&sY[m][kb * 8] = src[m * 48 + kb];
    }
  }
  __syncthreads();

  v16h afr[12];
#pragma unroll
  for (int kk = 0; kk < 12; ++kk) afr[kk] = load_a(&sY[0][0], 392, lane, kk * 32);

  for (int tile = wave; tile < 12; tile += 8) {
    const int eBase = tile * 16;
    v16h bq[4];
#pragma unroll
    for (int i = 0; i < 4; ++i)
      bq[i] = load_bh(w_out, DINC, eBase, i * 32, lane, DMODEL);
    v8f acc = {};
#pragma unroll
    for (int kk = 0; kk < 12; ++kk) {
      acc = __builtin_amdgcn_wmma_f32_16x16x32_f16(false, afr[kk], false, bq[kk & 3],
                                                   (short)0, acc, false, false);
      if (kk + 4 < 12)
        bq[kk & 3] = load_bh(w_out, DINC, eBase, (kk + 4) * 32, lane, DMODEL);
    }
    const int n = lane & 15, mB = (lane >> 4) << 3;
#pragma unroll
    for (int r = 0; r < 8; ++r)
      out[(size_t)(tok0 + mB + r) * DMODEL + eBase + n] = acc[r];
  }
}

extern "C" void kernel_launch(void* const* d_in, const int* in_sizes, int n_in,
                              void* d_out, int out_size, void* d_ws, size_t ws_size,
                              hipStream_t stream) {
  (void)in_sizes; (void)n_in; (void)out_size; (void)ws_size;
  const float* x         = (const float*)d_in[0];
  const float* norm_w    = (const float*)d_in[1];
  const float* norm_b    = (const float*)d_in[2];
  const float* in_proj_w = (const float*)d_in[3];
  const float* conv_w    = (const float*)d_in[4];
  const float* conv_b    = (const float*)d_in[5];
  const float* x_proj_w  = (const float*)d_in[6];
  const float* dt_proj_w = (const float*)d_in[7];
  const float* dt_proj_b = (const float*)d_in[8];
  const float* A_log     = (const float*)d_in[9];
  const float* Dskip     = (const float*)d_in[10];
  const float* out_proj_w= (const float*)d_in[11];
  float* out = (float*)d_out;

  char* base = (char*)d_ws;
  size_t off = 0;
  auto carve = [&](size_t bytes) -> void* {
    void* q = base + off;
    off += (bytes + 255) & ~(size_t)255;
    return q;
  };
  f16*   u_f16  = (f16*)carve((size_t)NTOK * DINC * sizeof(f16));
  f16*   zsil   = (f16*)carve((size_t)NTOK * DINC * sizeof(f16));
  f16*   uc     = (f16*)carve((size_t)NTOK * DINC * sizeof(f16));
  float* dbc    = (float*)carve((size_t)NTOK * 48 * sizeof(float));
  float* dtf    = (float*)carve((size_t)NTOK * DINC * sizeof(float));
  float* hloc   = (float*)carve((size_t)BB * DINC * NCH * NST * sizeof(float));
  float* h0     = (float*)carve((size_t)BB * DINC * NCH * NST * sizeof(float));
  float* dts    = (float*)carve((size_t)BB * DINC * NCH * sizeof(float));
  f16*   ybuf   = (f16*)carve((size_t)NTOK * DINC * sizeof(f16));
  f16*   w_in_h = (f16*)carve((size_t)2 * DINC * DMODEL * sizeof(f16));
  f16*   w_x_h  = (f16*)carve((size_t)(RTR + 2 * NST) * DINC * sizeof(f16));
  f16*   w_out_h= (f16*)carve((size_t)DMODEL * DINC * sizeof(f16));

  const dim3 blk(256);
  const int elemBlocks = (int)(((size_t)NTOK * DINC + 255) / 256);  // 49152

  const int nWin  = 2 * DINC * DMODEL;          // 147456
  const int nWx   = (RTR + 2 * NST) * DINC;     // 16896
  const int nWout = DMODEL * DINC;              // 73728
  k_cvt<<<dim3((nWin  + 255) / 256), blk, 0, stream>>>(in_proj_w,  w_in_h,  nWin);
  k_cvt<<<dim3((nWx   + 255) / 256), blk, 0, stream>>>(x_proj_w,   w_x_h,   nWx);
  k_cvt<<<dim3((nWout + 255) / 256), blk, 0, stream>>>(out_proj_w, w_out_h, nWout);

  k_ln_inproj<<<dim3(NTOK / 16), blk, 0, stream>>>(x, norm_w, norm_b, w_in_h,
                                                   u_f16, zsil);
  k_conv_silu<<<dim3(elemBlocks), blk, 0, stream>>>(u_f16, conv_w, conv_b, uc);
  k_xproj_dt<<<dim3(NTOK / 16), blk, 0, stream>>>(uc, w_x_h, dt_proj_w, dt_proj_b,
                                                  dbc, dtf);
  k_scan_local<<<dim3(BB * NCH * (DINC / 16)), blk, 0, stream>>>(dtf, uc, dbc, A_log,
                                                                 hloc, dts);
  k_scan_carry<<<dim3(BB * DINC / 16), blk, 0, stream>>>(hloc, dts, A_log, h0);
  k_scan_final<<<dim3(BB * NCH * (DINC / 16)), blk, 0, stream>>>(dtf, uc, dbc, A_log,
                                                                 Dskip, zsil, h0, ybuf);
  k_outproj<<<dim3(NTOK / 16), blk, 0, stream>>>(ybuf, w_out_h, out);
}
